// GATv2Net_16913581212034
// MI455X (gfx1250) — matrix operand order repaired and checked by reading, not compiled, not run on hardware
//
#include <hip/hip_runtime.h>

typedef __attribute__((ext_vector_type(16))) __bf16 v16bf;
typedef __attribute__((ext_vector_type(8)))  __bf16 v8bf;
typedef __attribute__((ext_vector_type(8)))  float  v8f;

#define GN  10000
#define GE  320000
#define GIN 256
#define GD  128
#define GH  4
#define GG  64
#define HD1 (GH * GD)   // 512
#define HD2 (GD)        // 128

// ---- ordered-uint encoding so atomicMax(u32) implements float max ----
__device__ __forceinline__ unsigned f2ord(float f) {
  unsigned u = __float_as_uint(f);
  return (u & 0x80000000u) ? ~u : (u | 0x80000000u);
}
__device__ __forceinline__ float ord2f(unsigned u) {
  return (u & 0x80000000u) ? __uint_as_float(u & 0x7fffffffu)
                           : __uint_as_float(~u);
}
// host-side: f2ord(-inf) == ~0xFF800000 == 0x007FFFFF
#define ENC_NEG_INF 0x007FFFFFu

// ---------------- utility kernels ----------------
__global__ void k_fill_f32(float* p, float v, long n) {
  long i = blockIdx.x * (long)blockDim.x + threadIdx.x;
  if (i < n) p[i] = v;
}
__global__ void k_fill_u32(unsigned* p, unsigned v, long n) {
  long i = blockIdx.x * (long)blockDim.x + threadIdx.x;
  if (i < n) p[i] = v;
}
__global__ void k_f32_to_bf16(const float* __restrict__ in, __bf16* __restrict__ out, long n) {
  long i = blockIdx.x * (long)blockDim.x + threadIdx.x;
  if (i < n) out[i] = (__bf16)in[i];
}
// in: [K][Ncols] row-major f32  ->  out: [Ncols][K] row-major bf16
__global__ void k_transpose_bf16(const float* __restrict__ in, __bf16* __restrict__ out,
                                 int K, int Ncols) {
  long i = blockIdx.x * (long)blockDim.x + threadIdx.x;
  if (i < (long)K * Ncols) {
    int k = (int)(i / Ncols), n = (int)(i % Ncols);
    out[(size_t)n * K + k] = (__bf16)in[i];
  }
}

// ---------------- bf16 WMMA GEMM: C[M,Ncols] = A[M,K] @ Bt[Ncols,K]^T + bias ----------------
// blockDim = (32, 2). Each wave computes a 16x64 strip (4 accumulators, A reused).
#define TPW 4
__global__ void k_gemm_bf16(const __bf16* __restrict__ A, const __bf16* __restrict__ Bt,
                            const float* __restrict__ bias, float* __restrict__ C,
                            int M, int K, int Ncols) {
  const int lane  = threadIdx.x;          // 0..31
  const int wave  = threadIdx.y;
  const int tileM = blockIdx.x * 16;
  const int tileN = (blockIdx.y * blockDim.y + wave) * (16 * TPW);
  const int lo16  = lane & 15;
  const int half  = lane >> 4;            // 0 or 1
  const int kA    = half * 8;             // A frag: K {0..7,16..23} / {8..15,24..31}
  const int kB    = half * 8;             // B frag: K {0..7,16..23} / {8..15,24..31}

  const __bf16* aPtr = A + (size_t)(tileM + lo16) * K;

  v8f acc[TPW] = {};
  for (int k0 = 0; k0 < K; k0 += 32) {
    // pull the k-block after next toward the WGP while this block's WMMAs run
    __builtin_prefetch(aPtr + k0 + 64, 0, 1);
    v8bf alo = *(const v8bf*)(aPtr + k0 + kA);
    v8bf ahi = *(const v8bf*)(aPtr + k0 + 16 + kA);
    v16bf a = __builtin_shufflevector(alo, ahi,
        0, 1, 2, 3, 4, 5, 6, 7, 8, 9, 10, 11, 12, 13, 14, 15);
#pragma unroll
    for (int t = 0; t < TPW; ++t) {
      const __bf16* bPtr = Bt + (size_t)(tileN + t * 16 + lo16) * K + k0 + kB;
      __builtin_prefetch(bPtr + 64, 0, 1);
      v8bf blo = *(const v8bf*)(bPtr);
      v8bf bhi = *(const v8bf*)(bPtr + 16);
      v16bf b = __builtin_shufflevector(blo, bhi,
          0, 1, 2, 3, 4, 5, 6, 7, 8, 9, 10, 11, 12, 13, 14, 15);
      acc[t] = __builtin_amdgcn_wmma_f32_16x16x32_bf16(
          false, a, false, b, (short)0, acc[t], false, false);
    }
  }
  // C/D layout: lanes 0-15 hold M = r (VGPR r), lanes 16-31 hold M = 8 + r; N = lane%16
#pragma unroll
  for (int t = 0; t < TPW; ++t) {
    int n = tileN + t * 16 + lo16;
    float bv = bias[n];
#pragma unroll
    for (int r = 0; r < 8; ++r) {
      int m = tileM + half * 8 + r;
      C[(size_t)m * Ncols + n] = acc[t][r] + bv;
    }
  }
}

// ---------------- edge kernels ----------------
// One wave per (e,h): lanes cooperatively read one contiguous 512B row of hs/hd
// (coalesced gather), reduce the attn dot product across the wave.
// score[e,h] = attn[h] . leaky_relu(hs[src[e],h,:] + hd[dst[e],h,:]); segment max into mEnc
__global__ void k_edge_score(const float* __restrict__ hs, const float* __restrict__ hd,
                             const float* __restrict__ attn,
                             const int* __restrict__ src, const int* __restrict__ dst,
                             float* __restrict__ score, unsigned* __restrict__ mEnc,
                             int E, int H, int D) {
  const int lane = threadIdx.x & 31;
  long wid = (blockIdx.x * (long)(blockDim.x >> 5)) + (threadIdx.x >> 5);
  if (wid >= (long)E * H) return;
  int e = (int)(wid / H), h = (int)(wid % H);
  int s = src[e], d = dst[e];
  // D = 128 floats per row -> 32 lanes x float4
  const float4* ps = (const float4*)(hs + ((size_t)s * H + h) * D);
  const float4* pd = (const float4*)(hd + ((size_t)d * H + h) * D);
  const float4* pa = (const float4*)(attn + (size_t)h * D);
  float4 a = ps[lane], b = pd[lane], w = pa[lane];
  float v0 = a.x + b.x; v0 = v0 > 0.f ? v0 : 0.2f * v0;
  float v1 = a.y + b.y; v1 = v1 > 0.f ? v1 : 0.2f * v1;
  float v2 = a.z + b.z; v2 = v2 > 0.f ? v2 : 0.2f * v2;
  float v3 = a.w + b.w; v3 = v3 > 0.f ? v3 : 0.2f * v3;
  float acc = v0 * w.x + v1 * w.y + v2 * w.z + v3 * w.w;
#pragma unroll
  for (int off = 16; off > 0; off >>= 1)
    acc += __shfl_down(acc, off, 32);
  if (lane == 0) {
    score[wid] = acc;
    atomicMax(mEnc + (size_t)d * H + h, f2ord(acc));
  }
}

// score -> exp(score - max); accumulate denominators
__global__ void k_edge_exp(float* __restrict__ score, const unsigned* __restrict__ mEnc,
                           float* __restrict__ denom, const int* __restrict__ dst,
                           int E, int H) {
  long idx = blockIdx.x * (long)blockDim.x + threadIdx.x;
  if (idx >= (long)E * H) return;
  int e = (int)(idx / H), h = (int)(idx % H);
  int d = dst[e];
  float m = ord2f(mEnc[(size_t)d * H + h]);
  float ex = __expf(score[idx] - m);
  score[idx] = ex;
  atomicAdd(denom + (size_t)d * H + h, ex);
}

// out[dst,h,:] += (ex/denom[dst,h]) * hs[src,h,:]
__global__ void k_edge_agg(const float* __restrict__ hs, const float* __restrict__ ex,
                           const float* __restrict__ denom,
                           const int* __restrict__ src, const int* __restrict__ dst,
                           float* __restrict__ out, int H, int D) {
  int e = blockIdx.x;
  int s = src[e], d = dst[e];
  int HD = H * D;
  for (int idx = threadIdx.x; idx < HD; idx += blockDim.x) {
    int h = idx / D;
    float alpha = ex[(size_t)e * H + h] / denom[(size_t)d * H + h];
    atomicAdd(out + (size_t)d * HD + idx, alpha * hs[(size_t)s * HD + idx]);
  }
}

// ---------------- graph readout ----------------
__global__ void k_pool(const float* __restrict__ h2, const int* __restrict__ gid,
                       float* __restrict__ hg, float* __restrict__ cnt, int N, int D) {
  long idx = blockIdx.x * (long)blockDim.x + threadIdx.x;
  if (idx >= (long)N * D) return;
  int i = (int)(idx / D), dd = (int)(idx % D);
  int g = gid[i];
  atomicAdd(hg + (size_t)g * D + dd, h2[idx]);
  if (dd == 0) atomicAdd(cnt + g, 1.0f);
}

__global__ void k_readout(const float* __restrict__ hg, const float* __restrict__ cnt,
                          const float* __restrict__ Wr1, const float* __restrict__ br1,
                          const float* __restrict__ Wr2, const float* __restrict__ br2,
                          float* __restrict__ out, int G, int D) {
  int g = threadIdx.x;
  if (g >= G) return;
  float inv = 1.0f / fmaxf(cnt[g], 1.0f);
  float acc = 0.f;
  for (int j = 0; j < 64; ++j) {
    float sj = br1[j];
    for (int dd = 0; dd < D; ++dd)
      sj += hg[(size_t)g * D + dd] * inv * Wr1[(size_t)dd * 64 + j];
    acc += fmaxf(sj, 0.f) * Wr2[j];
  }
  out[g] = acc + br2[0];
}

// ---------------- launcher ----------------
extern "C" void kernel_launch(void* const* d_in, const int* in_sizes, int n_in,
                              void* d_out, int out_size, void* d_ws, size_t ws_size,
                              hipStream_t stream) {
  (void)in_sizes; (void)n_in; (void)out_size; (void)ws_size;
  const float* x     = (const float*)d_in[0];
  const float* W1s   = (const float*)d_in[1];
  const float* b1s   = (const float*)d_in[2];
  const float* W1d   = (const float*)d_in[3];
  const float* b1d   = (const float*)d_in[4];
  const float* attn1 = (const float*)d_in[5];
  const float* W2s   = (const float*)d_in[6];
  const float* b2s   = (const float*)d_in[7];
  const float* W2d   = (const float*)d_in[8];
  const float* b2d   = (const float*)d_in[9];
  const float* attn2 = (const float*)d_in[10];
  const float* Wr1   = (const float*)d_in[11];
  const float* br1   = (const float*)d_in[12];
  const float* Wr2   = (const float*)d_in[13];
  const float* br2   = (const float*)d_in[14];
  const int*   src   = (const int*)d_in[15];
  const int*   dst   = (const int*)d_in[16];
  const int*   gid   = (const int*)d_in[17];
  float* out = (float*)d_out;

  char* p = (char*)d_ws;
  auto alloc = [&](size_t bytes) -> char* {
    char* r = p;
    p += (bytes + 255) & ~(size_t)255;
    return r;
  };

  __bf16*   xb   = (__bf16*)alloc((size_t)GN * GIN * 2);
  __bf16*   w1st = (__bf16*)alloc((size_t)HD1 * GIN * 2);
  __bf16*   w1dt = (__bf16*)alloc((size_t)HD1 * GIN * 2);
  float*    hs1  = (float*)alloc((size_t)GN * HD1 * 4);
  float*    hd1  = (float*)alloc((size_t)GN * HD1 * 4);   // reused as agg1 output
  float*    sc1  = (float*)alloc((size_t)GE * GH * 4);
  unsigned* m1   = (unsigned*)alloc((size_t)GN * GH * 4);
  float*    dn1  = (float*)alloc((size_t)GN * GH * 4);
  __bf16*   h1b  = (__bf16*)alloc((size_t)GN * HD1 * 2);
  __bf16*   w2st = (__bf16*)alloc((size_t)HD2 * HD1 * 2);
  __bf16*   w2dt = (__bf16*)alloc((size_t)HD2 * HD1 * 2);
  float*    hs2  = (float*)alloc((size_t)GN * HD2 * 4);
  float*    hd2  = (float*)alloc((size_t)GN * HD2 * 4);   // reused as agg2 output
  float*    sc2  = (float*)alloc((size_t)GE * 4);
  unsigned* m2   = (unsigned*)alloc((size_t)GN * 4);
  float*    dn2  = (float*)alloc((size_t)GN * 4);
  float*    hg   = (float*)alloc((size_t)GG * GD * 4);
  float*    cnt  = (float*)alloc((size_t)GG * 4);

  const int T = 256;
  auto blk = [](long n, int t) { return (unsigned)((n + t - 1) / t); };
  // 8 waves (of 32) per 256-thread block for the wave-per-(e,h) score kernel
  auto wblk = [](long nwaves) { return (unsigned)((nwaves + 7) / 8); };

  // ---- layer 1 projections (bf16 WMMA) ----
  k_f32_to_bf16<<<blk((long)GN * GIN, T), T, 0, stream>>>(x, xb, (long)GN * GIN);
  k_transpose_bf16<<<blk((long)GIN * HD1, T), T, 0, stream>>>(W1s, w1st, GIN, HD1);
  k_transpose_bf16<<<blk((long)GIN * HD1, T), T, 0, stream>>>(W1d, w1dt, GIN, HD1);
  dim3 gb(32, 2);
  dim3 gg1(GN / 16, HD1 / (16 * TPW * 2));  // (625, 4)
  k_gemm_bf16<<<gg1, gb, 0, stream>>>(xb, w1st, b1s, hs1, GN, GIN, HD1);
  k_gemm_bf16<<<gg1, gb, 0, stream>>>(xb, w1dt, b1d, hd1, GN, GIN, HD1);

  // ---- layer 1 edge softmax + aggregation ----
  k_fill_u32<<<blk((long)GN * GH, T), T, 0, stream>>>(m1, ENC_NEG_INF, (long)GN * GH);
  k_fill_f32<<<blk((long)GN * GH, T), T, 0, stream>>>(dn1, 0.f, (long)GN * GH);
  k_edge_score<<<wblk((long)GE * GH), T, 0, stream>>>(hs1, hd1, attn1, src, dst, sc1, m1, GE, GH, GD);
  k_edge_exp<<<blk((long)GE * GH, T), T, 0, stream>>>(sc1, m1, dn1, dst, GE, GH);
  k_fill_f32<<<blk((long)GN * HD1, T), T, 0, stream>>>(hd1, 0.f, (long)GN * HD1); // agg1 := 0 (hd1 no longer needed)
  k_edge_agg<<<GE, 256, 0, stream>>>(hs1, sc1, dn1, src, dst, hd1, GH, GD);

  // ---- layer 2 projections (bf16 WMMA) ----
  k_f32_to_bf16<<<blk((long)GN * HD1, T), T, 0, stream>>>(hd1, h1b, (long)GN * HD1);
  k_transpose_bf16<<<blk((long)HD1 * HD2, T), T, 0, stream>>>(W2s, w2st, HD1, HD2);
  k_transpose_bf16<<<blk((long)HD1 * HD2, T), T, 0, stream>>>(W2d, w2dt, HD1, HD2);
  dim3 gg2(GN / 16, HD2 / (16 * TPW * 2));  // (625, 1)
  k_gemm_bf16<<<gg2, gb, 0, stream>>>(h1b, w2st, b2s, hs2, GN, HD1, HD2);
  k_gemm_bf16<<<gg2, gb, 0, stream>>>(h1b, w2dt, b2d, hd2, GN, HD1, HD2);

  // ---- layer 2 edge softmax + aggregation (H = 1) ----
  k_fill_u32<<<blk((long)GN, T), T, 0, stream>>>(m2, ENC_NEG_INF, (long)GN);
  k_fill_f32<<<blk((long)GN, T), T, 0, stream>>>(dn2, 0.f, (long)GN);
  k_edge_score<<<wblk((long)GE), T, 0, stream>>>(hs2, hd2, attn2, src, dst, sc2, m2, GE, 1, GD);
  k_edge_exp<<<blk((long)GE, T), T, 0, stream>>>(sc2, m2, dn2, dst, GE, 1);
  k_fill_f32<<<blk((long)GN * HD2, T), T, 0, stream>>>(hd2, 0.f, (long)GN * HD2); // agg2 := 0
  k_edge_agg<<<GE, 128, 0, stream>>>(hs2, sc2, dn2, src, dst, hd2, 1, GD);

  // ---- graph mean-pool + MLP readout ----
  k_fill_f32<<<blk((long)GG * GD, T), T, 0, stream>>>(hg, 0.f, (long)GG * GD);
  k_fill_f32<<<1, GG, 0, stream>>>(cnt, 0.f, (long)GG);
  k_pool<<<blk((long)GN * GD, T), T, 0, stream>>>(hd2, gid, hg, cnt, GN, GD);
  k_readout<<<1, 64, 0, stream>>>(hg, cnt, Wr1, br1, Wr2, br2, out, GG, GD);
}
